// MLA_25537875542559
// MI455X (gfx1250) — matrix operand order repaired and checked by reading, not compiled, not run on hardware
//
#include <hip/hip_runtime.h>
#include <hip/hip_bf16.h>
#include <stdint.h>

// ---------------- problem constants ----------------
#define B_      2
#define S_      2048
#define DIM_    2048
#define H_      16
#define D_NOPE  128
#define D_ROPE  64
#define D_V     128
#define C_KV    512
#define QK_HEAD 192          // D_NOPE + D_ROPE
#define QC      576          // q_abs(512) + q_pe(64), fused per (b,s,h)
#define KC      576          // kv_n(512) + k_pe(64), fused per (b,s)

static constexpr double MSC_D = 0.1 * 3.6888794541139363 + 1.0;          // 0.1*ln(40)+1
static constexpr float  SOFTMAX_SCALE =
    (float)((MSC_D * MSC_D) / 13.856406460551018);                       // 192^-0.5 * mscale^2

// ---------------- CDNA5 WMMA types ----------------
typedef __attribute__((ext_vector_type(16))) __bf16 v16bf;
typedef __attribute__((ext_vector_type(8)))  float  v8f;
typedef __attribute__((ext_vector_type(4)))  unsigned v4u;
typedef __attribute__((ext_vector_type(4)))  int      v4i;

__device__ __forceinline__ uint16_t f2bf(float x) {
  unsigned u = __float_as_uint(x);
  unsigned r = u + 0x7FFFu + ((u >> 16) & 1u);   // round-to-nearest-even
  return (uint16_t)(r >> 16);
}

union FragU { v16bf v; v4u q[2]; };

// ---- async global->LDS staging (CDNA5) ----
// Probe-confirmed: builtin exists; param0 is v4i in AS1 (global), param1 v4i in AS3 (LDS).
#if defined(__has_builtin)
#  if __has_builtin(__builtin_amdgcn_global_load_async_to_lds_b128)
#    define HAVE_ASYNC_LDS 1
#  endif
#  if __has_builtin(__builtin_amdgcn_s_wait_asynccnt)
#    define HAVE_WAIT_ASYNC 1
#  endif
#endif

typedef __attribute__((address_space(1))) v4i as1_v4i;
typedef __attribute__((address_space(3))) v4i as3_v4i;

__device__ __forceinline__ void cp16_g2lds(uint16_t* lds, const uint16_t* g) {
#if defined(HAVE_ASYNC_LDS)
  __builtin_amdgcn_global_load_async_to_lds_b128(
      (as1_v4i*)(uintptr_t)g,
      (as3_v4i*)(uintptr_t)(unsigned)(uintptr_t)lds,
      0, 0);
#else
  *(v4u*)lds = *(const v4u*)g;
#endif
}
__device__ __forceinline__ void wait_async_cp() {
#if defined(HAVE_ASYNC_LDS)
#  if defined(HAVE_WAIT_ASYNC)
  __builtin_amdgcn_s_wait_asynccnt(0);
#  else
  asm volatile("s_wait_asynccnt 0" ::: "memory");
#  endif
#endif
}

// A fragment: 16x32 bf16 from row-major (M x K); K is lane-contiguous -> 2x b128.
// Layout (ISA 7.12.2): lane%16 = M; lanes>=16 shift K by 8; VGPRs 0-3 K=+0..7, 4-7 K=+16..23.
__device__ __forceinline__ v16bf load_a(const uint16_t* __restrict__ A, int lda,
                                        int k0, int lane) {
  int m = lane & 15, half = lane >> 4;
  const uint16_t* p = A + (size_t)m * lda + k0 + half * 8;
  FragU f;
  f.q[0] = *(const v4u*)p;
  f.q[1] = *(const v4u*)(p + 16);
  return f.v;
}

// B fragment: 32x16 where operand is W (N x K) row-major (B = W^T); 2x b128.
// Layout: lane%16 = N; lanes>=16 shift K by 8; VGPRs 0-3 K=+0..7, 4-7 K=+16..23.
__device__ __forceinline__ v16bf load_b_nt(const uint16_t* __restrict__ W, int ldw,
                                           int k0, int lane) {
  int n = lane & 15, half = lane >> 4;
  const uint16_t* p = W + (size_t)n * ldw + k0 + half * 8;
  FragU f;
  f.q[0] = *(const v4u*)p;
  f.q[1] = *(const v4u*)(p + 16);
  return f.v;
}

__device__ __forceinline__ v8f wmma_bf16(v16bf a, v16bf b, v8f c) {
  return __builtin_amdgcn_wmma_f32_16x16x32_bf16(false, a, false, b, (short)0, c,
                                                 false, false);
}

// ---------------- elementwise: fp32 -> bf16 ----------------
__global__ void cvt_bf16(const float* __restrict__ in, uint16_t* __restrict__ out,
                         size_t n) {
  size_t i = (size_t)blockIdx.x * blockDim.x + threadIdx.x;
  size_t stride = (size_t)gridDim.x * blockDim.x;
  for (; i < n; i += stride) out[i] = f2bf(in[i]);
}

// wb1 (H,256,512)[:, :128, :] -> wb1T (H,512,128): out[h][c][d] = in[h][d][c]
__global__ void cvt_wb1T(const float* __restrict__ in, uint16_t* __restrict__ out) {
  size_t n = (size_t)H_ * C_KV * D_NOPE;
  size_t i = (size_t)blockIdx.x * blockDim.x + threadIdx.x;
  size_t stride = (size_t)gridDim.x * blockDim.x;
  for (; i < n; i += stride) {
    int d = (int)(i & (D_NOPE - 1));
    int c = (int)((i >> 7) & (C_KV - 1));
    int h = (int)(i >> 16);
    out[i] = f2bf(in[((size_t)(h * 256 + d) << 9) + c]);
  }
}

// ---------------- generic WMMA GEMM: C[M,N] = A[M,K] @ W[N,K]^T ----------------
// One 16x64 tile per wave (1 A-frag : 4 B-frags : 4 WMMAs per k-step), 8 waves/block.
template <bool OBF>
__global__ __launch_bounds__(256) void gemm_wmma(
    const uint16_t* __restrict__ A, int lda, long aStep,
    const uint16_t* __restrict__ W, int ldw, long bStep,
    void* __restrict__ Cv, int ldc, long cStep,
    int M, int N, int K) {
  int z = blockIdx.z;
  A += (size_t)z * aStep;
  W += (size_t)z * bStep;
  int lane = threadIdx.x & 31, wave = threadIdx.x >> 5;
  int tilesN = N >> 6;
  int tile = blockIdx.x * 8 + wave;
  if (tile >= (M >> 4) * tilesN) return;        // wave-uniform: EXEC stays all-1s
  int tm = tile / tilesN, tn = tile % tilesN;
  const uint16_t* Arow = A + (size_t)(tm << 4) * lda;
  v8f acc[4];
#pragma unroll
  for (int i = 0; i < 4; ++i)
#pragma unroll
    for (int r = 0; r < 8; ++r) acc[i][r] = 0.f;
  for (int k0 = 0; k0 < K; k0 += 32) {
    v16bf a = load_a(Arow, lda, k0, lane);
#pragma unroll
    for (int i = 0; i < 4; ++i) {
      v16bf b = load_b_nt(W + (size_t)((tn << 6) + i * 16) * ldw, ldw, k0, lane);
      acc[i] = wmma_bf16(a, b, acc[i]);
    }
  }
  int n = lane & 15, half = lane >> 4;
#pragma unroll
  for (int i = 0; i < 4; ++i) {
    int col = (tn << 6) + i * 16 + n;
    if (OBF) {
      uint16_t* C = (uint16_t*)Cv + (size_t)z * cStep;
#pragma unroll
      for (int r = 0; r < 8; ++r)
        C[(size_t)((tm << 4) + r + half * 8) * ldc + col] = f2bf(acc[i][r]);
    } else {
      float* C = (float*)Cv + (size_t)z * cStep;
#pragma unroll
      for (int r = 0; r < 8; ++r)
        C[(size_t)((tm << 4) + r + half * 8) * ldc + col] = acc[i][r];
    }
  }
}

// ---------------- RoPE on q_pe, pack q_nope/q_pe ----------------
__global__ __launch_bounds__(128) void rope_q_kernel(
    const float* __restrict__ q, const float* __restrict__ fc,
    const float* __restrict__ fs, uint16_t* __restrict__ qnope,
    uint16_t* __restrict__ qcat) {
  int idx = blockIdx.x;                   // (b*S + s)*H + h
  int s = (idx / H_) % S_;
  int t = threadIdx.x;
  const float* row = q + (size_t)idx * QK_HEAD;
  qnope[(size_t)idx * D_NOPE + t] = f2bf(row[t]);
  if (t < D_ROPE / 2) {
    float c = fc[s * (D_ROPE / 2) + t], sn = fs[s * (D_ROPE / 2) + t];
    float x0 = row[D_NOPE + 2 * t], x1 = row[D_NOPE + 2 * t + 1];
    uint16_t* o = qcat + (size_t)idx * QC + C_KV;
    o[2 * t]     = f2bf(x0 * c - x1 * sn);
    o[2 * t + 1] = f2bf(x0 * sn + x1 * c);
  }
}

// ---------------- RMSNorm(kv)+RoPE(k_pe) -> kcat ----------------
__global__ __launch_bounds__(128) void kv_norm_rope_kernel(
    const float* __restrict__ kvf, const float* __restrict__ w,
    const float* __restrict__ fc, const float* __restrict__ fs,
    uint16_t* __restrict__ kcat) {
  __shared__ float red[128];
  int bs = blockIdx.x;
  int s = bs % S_;
  int t = threadIdx.x;
  const float* row = kvf + (size_t)bs * (C_KV + D_ROPE);
  float p = 0.f;
#pragma unroll
  for (int j = 0; j < 4; ++j) { float v = row[t + j * 128]; p += v * v; }
  red[t] = p;
  __syncthreads();
  for (int o = 64; o > 0; o >>= 1) {
    if (t < o) red[t] += red[t + o];
    __syncthreads();
  }
  float rms = rsqrtf(red[0] / (float)C_KV + 1e-6f);
  uint16_t* out = kcat + (size_t)bs * KC;
#pragma unroll
  for (int j = 0; j < 4; ++j) {
    int c = t + j * 128;
    out[c] = f2bf(row[c] * rms * w[c]);
  }
  if (t < D_ROPE / 2) {
    float c = fc[s * (D_ROPE / 2) + t], sn = fs[s * (D_ROPE / 2) + t];
    float x0 = row[C_KV + 2 * t], x1 = row[C_KV + 2 * t + 1];
    out[C_KV + 2 * t]     = f2bf(x0 * c - x1 * sn);
    out[C_KV + 2 * t + 1] = f2bf(x0 * sn + x1 * c);
  }
}

// ---------------- flash attention (one pass, WMMA, LDS-staged) ----------------
// grid=(S/32, B*H), block=256 (8 waves): 32 query rows of one (b,h).
// Per 64-wide t-chunk:
//   0a. async-copy kv chunk (64 x 576 bf16) global->LDS (coalesced b128)
//   0b. build LDS transpose vt[c][t] of the kv_n part (for P@V B-frags)
//   1.  8 waves: 16x16 score tiles, K=576 bf16 WMMA, B-frags from LDS
//   2.  wave0: online softmax (lane==row), bf16 P -> LDS
//   3.  all waves: rescale f32 acc, P@V (A from LDS P, B from vt), 16 WMMAs
static constexpr int AT_ROWS = 32, AT_TCH = 64;
static constexpr int LKV = KC;        // kvrow stride (elems); 1152B rows, 16B aligned
static constexpr int LVT = 72;        // vt stride: 144B rows, 16B aligned
static constexpr int LSC = 65;        // score stride (floats)
static constexpr int LDP = 72;        // P stride: 144B rows, 16B aligned
static constexpr size_t AT_SMEM =
    (size_t)AT_TCH * LKV * 2 + (size_t)C_KV * LVT * 2 +
    (size_t)AT_ROWS * LSC * 4 + (size_t)AT_ROWS * LDP * 2 + 3 * AT_ROWS * 4;

__global__ __launch_bounds__(256) void mla_flash_attn(
    const uint16_t* __restrict__ qcat, const uint16_t* __restrict__ kcat,
    uint16_t* __restrict__ obuf) {
  extern __shared__ __align__(16) char smem[];
  uint16_t* kvrow = (uint16_t*)smem;                       // [64][576]
  uint16_t* vt    = kvrow + AT_TCH * LKV;                  // [512][72]
  float*    s_sc  = (float*)(vt + C_KV * LVT);             // [32][65]
  uint16_t* s_p   = (uint16_t*)(s_sc + AT_ROWS * LSC);     // [32][72]
  float*    s_m   = (float*)(s_p + AT_ROWS * LDP);
  float*    s_l   = s_m + AT_ROWS;
  float*    s_al  = s_l + AT_ROWS;

  int bh = blockIdx.y, b = bh / H_, h = bh % H_;
  int s0 = blockIdx.x * AT_ROWS;
  int tid = threadIdx.x, wave = tid >> 5, lane = tid & 31;
  int n = lane & 15, half = lane >> 4;
  int rg = wave >> 2;          // row group (0..1): 16 rows
  int ct = wave & 3;           // stage1: t sub-tile / stage3: 128-col group

  const uint16_t* qA = qcat + ((size_t)b * S_ * H_ + h) * QC;
  const int ldq = H_ * QC;
  const uint16_t* kB = kcat + (size_t)b * S_ * KC;

  if (tid < AT_ROWS) { s_m[tid] = -INFINITY; s_l[tid] = 0.f; }

  v8f acc[8];
#pragma unroll
  for (int i = 0; i < 8; ++i)
#pragma unroll
    for (int r = 0; r < 8; ++r) acc[i][r] = 0.f;

  __syncthreads();

  const uint16_t* Arow = qA + (size_t)(s0 + rg * 16) * ldq;
  int nch = (s0 + AT_ROWS - 1) / AT_TCH + 1;     // chunks with t0 <= s_max (causal)
  for (int ch = 0; ch < nch; ++ch) {
    int t0 = ch * AT_TCH;
    // ---- stage 0a: stage kv chunk into LDS (async if available) ----
#pragma unroll
    for (int j = 0; j < 18; ++j) {
      int seg = tid + j * 256;                   // 4608 x 16B = 64*576*2 B
      int t = seg / 72, c16 = seg % 72;
      cp16_g2lds(&kvrow[t * LKV + c16 * 8],
                 kB + (size_t)(t0 + t) * KC + c16 * 8);
    }
    wait_async_cp();
    __syncthreads();
    // ---- stage 0b: transpose kv_n part into vt[c][t] ----
#pragma unroll
    for (int j = 0; j < 16; ++j) {
      int seg = tid + j * 256;                   // 64 rows x 64 segs (c<512)
      int t = seg >> 6, c8 = seg & 63;
      union { v4u q; uint16_t e[8]; } u;
      u.q = *(const v4u*)&kvrow[t * LKV + c8 * 8];
#pragma unroll
      for (int e = 0; e < 8; ++e) vt[(c8 * 8 + e) * LVT + t] = u.e[e];
    }
    __syncthreads();
    // ---- stage 1: scores (one 16x16 tile per wave, K=576) ----
    {
      const uint16_t* Wk = &kvrow[(ct * 16) * LKV];
      v8f sc = {0.f, 0.f, 0.f, 0.f, 0.f, 0.f, 0.f, 0.f};
      for (int k0 = 0; k0 < QC; k0 += 32) {
        v16bf a  = load_a(Arow, ldq, k0, lane);
        v16bf bf = load_b_nt(Wk, LKV, k0, lane);
        sc = wmma_bf16(a, bf, sc);
      }
      int tcol = t0 + ct * 16 + n;
#pragma unroll
      for (int r = 0; r < 8; ++r) {
        int srow = s0 + rg * 16 + r + half * 8;
        float v = sc[r] * SOFTMAX_SCALE;
        if (tcol > srow) v = -INFINITY;          // causal mask
        s_sc[(rg * 16 + r + half * 8) * LSC + ct * 16 + n] = v;
      }
    }
    __syncthreads();
    // ---- stage 2: online softmax (wave 0, lane == row) ----
    if (wave == 0) {
      float mo = s_m[lane], mx = mo;
#pragma unroll 8
      for (int c = 0; c < AT_TCH; ++c) mx = fmaxf(mx, s_sc[lane * LSC + c]);
      float al = __expf(mo - mx);                // 0 on first chunk (mo = -inf)
      float sum = 0.f;
#pragma unroll 8
      for (int c = 0; c < AT_TCH; ++c) {
        float pv = __expf(s_sc[lane * LSC + c] - mx);
        s_p[lane * LDP + c] = f2bf(pv);
        sum += pv;
      }
      s_l[lane]  = s_l[lane] * al + sum;
      s_m[lane]  = mx;
      s_al[lane] = al;
    }
    __syncthreads();
    // ---- stage 3: rescale acc, then o += P @ V (K=64, V from vt) ----
    {
      float al[8];
#pragma unroll
      for (int r = 0; r < 8; ++r) al[r] = s_al[rg * 16 + r + half * 8];
#pragma unroll
      for (int i = 0; i < 8; ++i)
#pragma unroll
        for (int r = 0; r < 8; ++r) acc[i][r] *= al[r];

      const uint16_t* Pb = &s_p[(rg * 16) * LDP];
      v16bf p0 = load_a(Pb, LDP, 0, lane);
      v16bf p1 = load_a(Pb, LDP, 32, lane);
#pragma unroll
      for (int i = 0; i < 8; ++i) {
        const uint16_t* Vb = &vt[(ct * 128 + i * 16) * LVT];
        v16bf b0 = load_b_nt(Vb, LVT, 0, lane);
        v16bf b1 = load_b_nt(Vb, LVT, 32, lane);
        acc[i] = wmma_bf16(p0, b0, acc[i]);
        acc[i] = wmma_bf16(p1, b1, acc[i]);
      }
    }
    __syncthreads();
  }
  // ---- epilogue: divide by l, store bf16 (B,S,H,512) ----
  float li[8];
#pragma unroll
  for (int r = 0; r < 8; ++r) li[r] = 1.f / s_l[rg * 16 + r + half * 8];
#pragma unroll
  for (int i = 0; i < 8; ++i)
#pragma unroll
    for (int r = 0; r < 8; ++r) {
      int srow = s0 + rg * 16 + r + half * 8;
      int c = ct * 128 + i * 16 + n;
      obuf[(((size_t)b * S_ + srow) * H_ + h) * C_KV + c] = f2bf(acc[i][r] * li[r]);
    }
}

// ---------------- workspace layout (bytes) ----------------
static constexpr size_t OFF_XBF   = 0;
static constexpr size_t OFF_WQBF  = OFF_XBF   + 16777216;
static constexpr size_t OFF_WKVA  = OFF_WQBF  + 12582912;
static constexpr size_t OFF_WKVB  = OFF_WKVA  + 2359296;
static constexpr size_t OFF_WOBF  = OFF_WKVB  + 4194304;
static constexpr size_t OFF_QF32  = OFF_WOBF  + 8388608;
static constexpr size_t OFF_KVF32 = OFF_QF32  + 50331648;
static constexpr size_t OFF_QNOPE = OFF_KVF32 + 9437184;
static constexpr size_t OFF_QCAT  = OFF_QNOPE + 16777216;
static constexpr size_t OFF_KCAT  = OFF_QCAT  + 75497472;
static constexpr size_t OFF_OBF   = OFF_KCAT  + 4718592;
static constexpr size_t OFF_ATTN  = OFF_OBF   + 67108864;
static constexpr size_t OFF_WB1T  = OFF_ATTN  + 16777216;   // H*512*128 bf16 = 2MB

extern "C" void kernel_launch(void* const* d_in, const int* in_sizes, int n_in,
                              void* d_out, int out_size, void* d_ws, size_t ws_size,
                              hipStream_t stream) {
  (void)in_sizes; (void)n_in; (void)out_size; (void)ws_size;
  const float* x    = (const float*)d_in[0];
  const float* fcos = (const float*)d_in[2];
  const float* fsin = (const float*)d_in[3];
  const float* wq   = (const float*)d_in[4];
  const float* wkva = (const float*)d_in[5];
  const float* kvw  = (const float*)d_in[6];
  const float* wkvb = (const float*)d_in[7];
  const float* wo   = (const float*)d_in[8];

  char* ws = (char*)d_ws;
  uint16_t* xbf    = (uint16_t*)(ws + OFF_XBF);
  uint16_t* wqbf   = (uint16_t*)(ws + OFF_WQBF);
  uint16_t* wkvabf = (uint16_t*)(ws + OFF_WKVA);
  uint16_t* wkvbbf = (uint16_t*)(ws + OFF_WKVB);
  uint16_t* wobf   = (uint16_t*)(ws + OFF_WOBF);
  float*    qf     = (float*)(ws + OFF_QF32);
  float*    kvf    = (float*)(ws + OFF_KVF32);
  uint16_t* qnope  = (uint16_t*)(ws + OFF_QNOPE);
  uint16_t* qcat   = (uint16_t*)(ws + OFF_QCAT);
  uint16_t* kcat   = (uint16_t*)(ws + OFF_KCAT);
  uint16_t* obf    = (uint16_t*)(ws + OFF_OBF);
  uint16_t* attnbf = (uint16_t*)(ws + OFF_ATTN);
  uint16_t* wb1T   = (uint16_t*)(ws + OFF_WB1T);

  // 1) convert to bf16 (+ transposed wb1 for the q_abs GEMM)
  cvt_bf16<<<2048, 256, 0, stream>>>(x,    xbf,    (size_t)B_ * S_ * DIM_);
  cvt_bf16<<<2048, 256, 0, stream>>>(wq,   wqbf,   (size_t)H_ * QK_HEAD * DIM_);
  cvt_bf16<<<1024, 256, 0, stream>>>(wkva, wkvabf, (size_t)(C_KV + D_ROPE) * DIM_);
  cvt_bf16<<<1024, 256, 0, stream>>>(wkvb, wkvbbf, (size_t)H_ * 256 * C_KV);
  cvt_bf16<<<2048, 256, 0, stream>>>(wo,   wobf,   (size_t)DIM_ * DIM_);
  cvt_wb1T<<<1024, 256, 0, stream>>>(wkvb, wb1T);

  const int M = B_ * S_;                       // 4096 token rows
  // 2) q = x @ wq^T  (4096 x 3072, K=2048): tiles = 256*48 -> 1536 blocks
  gemm_wmma<false><<<dim3(1536, 1, 1), 256, 0, stream>>>(
      xbf, DIM_, 0, wqbf, DIM_, 0, qf, H_ * QK_HEAD, 0, M, H_ * QK_HEAD, DIM_);
  // 3) kv = x @ wkv_a^T  (4096 x 576, K=2048): tiles = 256*9 -> 288 blocks
  gemm_wmma<false><<<dim3(288, 1, 1), 256, 0, stream>>>(
      xbf, DIM_, 0, wkvabf, DIM_, 0, kvf, C_KV + D_ROPE, 0, M, C_KV + D_ROPE, DIM_);
  // 4) rope(q_pe) -> qcat[...,512:576]; q_nope -> bf16
  rope_q_kernel<<<B_ * S_ * H_, 128, 0, stream>>>(qf, fcos, fsin, qnope, qcat);
  // 5) rmsnorm(kv)+rope(k_pe) -> kcat
  kv_norm_rope_kernel<<<B_ * S_, 128, 0, stream>>>(kvf, kvw, fcos, fsin, kcat);
  // 6) q_abs = q_nope @ wb1T^T per head -> qcat[...,0:512]  (K=128, N=512)
  gemm_wmma<true><<<dim3(256, 1, H_), 256, 0, stream>>>(
      qnope, H_ * D_NOPE, D_NOPE, wb1T, D_NOPE, (long)C_KV * D_NOPE,
      qcat, H_ * QC, QC, M, C_KV, D_NOPE);
  // 7) flash attention (dynamic LDS ~157KB -> 2 blocks/WGP)
  mla_flash_attn<<<dim3(S_ / 32, B_ * H_), 256, AT_SMEM, stream>>>(qcat, kcat, obf);
  // 8) attn = o @ wb2^T per head  (K=512, N=128): tiles = 256*2 -> 64 blocks
  gemm_wmma<true><<<dim3(64, 1, H_), 256, 0, stream>>>(
      obf, H_ * C_KV, C_KV, wkvbbf + (size_t)D_NOPE * C_KV, C_KV, 256L * C_KV,
      attnbf, H_ * D_V, D_V, M, D_V, C_KV);
  // 9) out = attn @ wo^T  (4096 x 2048, K=2048): tiles = 256*32 -> 1024 blocks
  gemm_wmma<false><<<dim3(1024, 1, 1), 256, 0, stream>>>(
      attnbf, H_ * D_V, 0, wobf, DIM_, 0, (float*)d_out, DIM_, 0, M, DIM_, DIM_);
}
